// GNNEncoder_29240137351458
// MI455X (gfx1250) — matrix operand order repaired and checked by reading, not compiled, not run on hardware
//
#include <hip/hip_runtime.h>
#include <hip/hip_bf16.h>
#include <math.h>

// ---------------------------------------------------------------------------
// GNN encoder forward for MI455X (gfx1250, wave32).
// GEMMs via v_wmma_f32_16x16x32_f16; segment ops via atomics; LN/attention
// via wave32 shuffle reductions.
//
// Input order (JAX pytree flatten, dict keys sorted):
//  0 x[N,64]  1 edge_index[2,E](int)  2 edge_attr[E,16]  3 batch[N](int)
//  4 atom_W[64,128] 5 atom_b[128]
//  6+13l (l=0..2): att_dst[4,32], att_src[4,32], eps[1], gat_W[128,128],
//                  gat_b[128], ln1_b, ln1_g, ln2_b, ln2_g,
//                  mlp_W1[128,128], mlp_W2[128,128], mlp_b1, mlp_b2
//  45 bond_W[16,128] 46 bond_b[128]
//  47 W_hh[1536,384] 48 W_ih[1536,768] 49 b_hh[1536] 50 b_ih[1536]
// ---------------------------------------------------------------------------

#define NN   100000
#define NE   400000
#define NG   4000
#define HID  128
#define DJK  384   // HID*LAYERS

typedef __attribute__((ext_vector_type(16))) _Float16 v16h;
typedef __attribute__((ext_vector_type(8)))  _Float16 v8h;
typedef __attribute__((ext_vector_type(8)))  float    v8f;

__device__ __forceinline__ float sigf(float x){ return 1.f/(1.f+expf(-x)); }
__device__ __forceinline__ float lrelu(float x){ return x > 0.f ? x : 0.2f*x; }
// order-preserving float<->uint map for atomicMax on floats
__device__ __forceinline__ unsigned f2o(float f){
  unsigned u = __float_as_uint(f);
  return (u & 0x80000000u) ? ~u : (u | 0x80000000u);
}
__device__ __forceinline__ float o2f(unsigned u){
  unsigned v = (u & 0x80000000u) ? (u & 0x7fffffffu) : ~u;
  return __uint_as_float(v);
}

// ---------------- weight packing into WMMA B-fragment layout ----------------
// frag[kc][ct][lane][i] (i=0..15 f16):  B[kc*32 + (lane/16)*16 + i][ct*16 + lane%16]
__global__ void k_pack_bfrag(const float* __restrict__ W, int K, int NC, int KP,
                             int ldsrc, int transposed, _Float16* __restrict__ frag,
                             int total){
  int t = blockIdx.x*256 + threadIdx.x;
  if (t >= total) return;
  int i    = t & 15;
  int lane = (t >> 4) & 31;
  int nct  = NC >> 4;
  int ct   = (t >> 9) % nct;
  int kc   = t / (nct << 9);
  int k = kc*32 + ((lane >> 4) << 4) + i;
  int n = ct*16 + (lane & 15);
  float v = 0.f;
  if (k < K) v = transposed ? W[(size_t)n*ldsrc + k] : W[(size_t)k*ldsrc + n];
  frag[t] = (_Float16)v;
}

// ---------------- WMMA GEMM: C[R,NC] (+)= act(A[R,K] * B + bias) ------------
// Block: 256 thr (8 waves) -> 64 rows x 128 cols; grid (ceil(R/64), NC/128).
// K must be a multiple of 8 (true for 16/64/128/384/768 used here).
__global__ void k_wmma_gemm(const float* __restrict__ A, int lda, int R, int K, int KP,
                            const _Float16* __restrict__ Bfrag, int NC,
                            const float* __restrict__ bias, float* __restrict__ C,
                            int ldc, int act, int accum){
  __shared__ __align__(16) _Float16 Alds[64*40];  // padded stride 40 halves (80B rows)
  const int wave = threadIdx.x >> 5, lane = threadIdx.x & 31;
  const int row0 = blockIdx.x * 64;
  const int ct   = blockIdx.y * 8 + wave;   // 16-col tile index
  v8f zero = {0.f,0.f,0.f,0.f,0.f,0.f,0.f,0.f};
  v8f acc[4];
#pragma unroll
  for (int r = 0; r < 4; ++r) acc[r] = zero;

  const int sr = threadIdx.x >> 2;          // staging row 0..63
  const int sc = (threadIdx.x & 3) << 3;    // staging col base 0/8/16/24
  // clamp row: out-of-range rows only feed tile rows the epilogue never stores
  int srow = row0 + sr; if (srow >= R) srow = R - 1;
  const float* arow = A + (size_t)srow*lda;
  _Float16* sp = Alds + sr*40 + sc;         // 16B-aligned (80B rows, sc*2 in {0,16,32,48})
  const int nkc = KP >> 5;
  const int m  = lane & 15;
  const int g8 = (lane >> 4) << 3;

  for (int kc = 0; kc < nkc; ++kc){
    // branch-free staging: 2x global b128 + cvt + 1x LDS b128 store
    int kb = kc*32 + sc;
    float msk = (kb + 8 <= K) ? 1.f : 0.f;
    int kbc = (kb + 8 <= K) ? kb : 0;
    float4 va = *(const float4*)(arow + kbc);
    float4 vb = *(const float4*)(arow + kbc + 4);
    v8h hv;
    hv[0] = (_Float16)(va.x*msk); hv[1] = (_Float16)(va.y*msk);
    hv[2] = (_Float16)(va.z*msk); hv[3] = (_Float16)(va.w*msk);
    hv[4] = (_Float16)(vb.x*msk); hv[5] = (_Float16)(vb.y*msk);
    hv[6] = (_Float16)(vb.z*msk); hv[7] = (_Float16)(vb.w*msk);
    *(v8h*)sp = hv;
    __syncthreads();
    v16h bf = *((const v16h*)Bfrag + ((size_t)kc*(NC >> 4) + ct)*32 + lane);
#pragma unroll
    for (int rt = 0; rt < 4; ++rt){
      const _Float16* ap = &Alds[(rt*16 + m)*40];
      v16h af;
#pragma unroll
      for (int i = 0; i < 8; ++i){ af[i] = ap[2*g8 + i]; af[i + 8] = ap[2*g8 + 8 + i]; }
      acc[rt] = __builtin_amdgcn_wmma_f32_16x16x32_f16(
          false, af, false, bf, (short)0, acc[rt], false, false);
    }
    __syncthreads();
  }
  const int cn = ct*16 + (lane & 15);
  const float bv = bias ? bias[cn] : 0.f;
  const int rbase = row0 + ((lane >> 4) << 3);
  if (row0 + 64 <= R){                       // fast path: full tile, no guards
#pragma unroll
    for (int rt = 0; rt < 4; ++rt){
#pragma unroll
      for (int v = 0; v < 8; ++v){
        float x = acc[rt][v] + bv;
        float* p = C + (size_t)(rbase + rt*16 + v)*ldc + cn;
        if (accum) x += *p;
        if (act)   x = fmaxf(x, 0.f);
        *p = x;
      }
    }
  } else {
#pragma unroll
    for (int rt = 0; rt < 4; ++rt){
#pragma unroll
      for (int v = 0; v < 8; ++v){
        int gr = rbase + rt*16 + v;
        if (gr < R){
          float x = acc[rt][v] + bv;
          float* p = C + (size_t)gr*ldc + cn;
          if (accum) x += *p;
          if (act)   x = fmaxf(x, 0.f);
          *p = x;
        }
      }
    }
  }
}

// ---------------- elementwise / fill ----------------------------------------
__global__ void k_fill_f32(float* p, float v, int n){
  int t = blockIdx.x*256 + threadIdx.x; if (t < n) p[t] = v;
}
__global__ void k_fill_u32(unsigned* p, unsigned v, int n){
  int t = blockIdx.x*256 + threadIdx.x; if (t < n) p[t] = v;
}
__global__ void k_addvec(const float* a, const float* b, float* o, int n){
  int t = blockIdx.x*256 + threadIdx.x; if (t < n) o[t] = a[t] + b[t];
}
__global__ void k_scale(const float* __restrict__ h, const float* __restrict__ epsp,
                        float* __restrict__ o, int n){
  int t = blockIdx.x*256 + threadIdx.x; if (t < n) o[t] = (1.f + *epsp) * h[t];
}
__global__ void k_copy(const float* a, float* o, int n){
  int t = blockIdx.x*256 + threadIdx.x; if (t < n) o[t] = a[t];
}

// ---------------- GINE message + aggregation (wave per edge) ----------------
__global__ void k_gine_msg(const int* __restrict__ src, const int* __restrict__ dst,
                           const float* __restrict__ h, const float* __restrict__ ea,
                           float* __restrict__ g1){
  int t = blockIdx.x*256 + threadIdx.x;
  int e = t >> 5, lane = t & 31;
  if (e >= NE) return;
  int s = src[e], d = dst[e];
  float4 hv = *(const float4*)(h  + (size_t)s*HID + lane*4);
  float4 ev = *(const float4*)(ea + (size_t)e*HID + lane*4);
  float* o = g1 + (size_t)d*HID + lane*4;
  atomicAdd(o + 0, fmaxf(hv.x + ev.x, 0.f));
  atomicAdd(o + 1, fmaxf(hv.y + ev.y, 0.f));
  atomicAdd(o + 2, fmaxf(hv.z + ev.z, 0.f));
  atomicAdd(o + 3, fmaxf(hv.w + ev.w, 0.f));
}

// ---------------- LayerNorm epilogue (wave per node, 128 feats) -------------
// out = LN(relu(V[/den + bias])) * g + b + res
__global__ void k_ln(const float* __restrict__ V, const float* __restrict__ den,
                     const float* __restrict__ bias, const float* __restrict__ g,
                     const float* __restrict__ b, const float* __restrict__ res,
                     float* __restrict__ out, int n){
  int t = blockIdx.x*256 + threadIdx.x;
  int node = t >> 5, lane = t & 31;
  if (node >= n) return;
  int f = lane*4;
  float4 v = *(const float4*)(V + (size_t)node*HID + f);
  if (den){
    float inv = 1.f / den[node*4 + (lane >> 3)];
    v.x *= inv; v.y *= inv; v.z *= inv; v.w *= inv;
  }
  if (bias){
    float4 bb = *(const float4*)(bias + f);
    v.x += bb.x; v.y += bb.y; v.z += bb.z; v.w += bb.w;
  }
  v.x = fmaxf(v.x, 0.f); v.y = fmaxf(v.y, 0.f);
  v.z = fmaxf(v.z, 0.f); v.w = fmaxf(v.w, 0.f);
  float s  = v.x + v.y + v.z + v.w;
  float ss = v.x*v.x + v.y*v.y + v.z*v.z + v.w*v.w;
#pragma unroll
  for (int msk = 1; msk < 32; msk <<= 1){ s += __shfl_xor(s, msk); ss += __shfl_xor(ss, msk); }
  float mu = s * (1.f/HID);
  float var = ss * (1.f/HID) - mu*mu;
  float rstd = rsqrtf(var + 1e-5f);
  float4 gg = *(const float4*)(g + f);
  float4 b2 = *(const float4*)(b + f);
  float4 rr = *(const float4*)(res + (size_t)node*HID + f);
  float4 o;
  o.x = (v.x - mu)*rstd*gg.x + b2.x + rr.x;
  o.y = (v.y - mu)*rstd*gg.y + b2.y + rr.y;
  o.z = (v.z - mu)*rstd*gg.z + b2.z + rr.z;
  o.w = (v.w - mu)*rstd*gg.w + b2.w + rr.w;
  *(float4*)(out + (size_t)node*HID + f) = o;
}

// ---------------- GAT attention --------------------------------------------
__global__ void k_att(const float* __restrict__ xw, const float* __restrict__ aS,
                      const float* __restrict__ aD, float* __restrict__ asrc,
                      float* __restrict__ adst){
  int t = blockIdx.x*256 + threadIdx.x;
  int node = t >> 5, lane = t & 31;
  if (node >= NN) return;
  int f = lane*4;
  float4 x = *(const float4*)(xw + (size_t)node*HID + f);
  float4 s4 = *(const float4*)(aS + f);
  float4 d4 = *(const float4*)(aD + f);
  float ps = x.x*s4.x + x.y*s4.y + x.z*s4.z + x.w*s4.w;
  float pd = x.x*d4.x + x.y*d4.y + x.z*d4.z + x.w*d4.w;
#pragma unroll
  for (int msk = 1; msk < 8; msk <<= 1){ ps += __shfl_xor(ps, msk); pd += __shfl_xor(pd, msk); }
  if ((lane & 7) == 0){
    int hd = lane >> 3;
    asrc[node*4 + hd] = ps;
    adst[node*4 + hd] = pd;
  }
}
__global__ void k_gat_maxinit(const float* asrc, const float* adst, unsigned* m, int n4){
  int t = blockIdx.x*256 + threadIdx.x; if (t >= n4) return;
  m[t] = f2o(lrelu(asrc[t] + adst[t]));               // self-loop score
}
__global__ void k_gat_edge_max(const int* src, const int* dst, const float* asrc,
                               const float* adst, unsigned* m){
  int t = blockIdx.x*256 + threadIdx.x; if (t >= NE*4) return;
  int e = t >> 2, hd = t & 3;
  int s = src[e], d = dst[e];
  float sc = lrelu(asrc[s*4 + hd] + adst[d*4 + hd]);
  atomicMax(&m[d*4 + hd], f2o(sc));
}
__global__ void k_gat_self(const float* __restrict__ xw, const float* asrc,
                           const float* adst, const unsigned* m,
                           float* __restrict__ den, float* __restrict__ outb){
  int t = blockIdx.x*256 + threadIdx.x;
  int node = t >> 5, lane = t & 31;
  if (node >= NN) return;
  int hd = lane >> 3, f = lane*4;
  float sc = lrelu(asrc[node*4 + hd] + adst[node*4 + hd]);
  float e_ = expf(sc - o2f(m[node*4 + hd]));
  if ((lane & 7) == 0) den[node*4 + hd] = e_;
  float4 x = *(const float4*)(xw + (size_t)node*HID + f);
  float4 o; o.x = e_*x.x; o.y = e_*x.y; o.z = e_*x.z; o.w = e_*x.w;
  *(float4*)(outb + (size_t)node*HID + f) = o;
}
__global__ void k_gat_edge_acc(const int* src, const int* dst, const float* asrc,
                               const float* adst, const unsigned* m,
                               const float* __restrict__ xw,
                               float* __restrict__ den, float* __restrict__ outb){
  int t = blockIdx.x*256 + threadIdx.x;
  int e = t >> 5, lane = t & 31;
  if (e >= NE) return;
  int s = src[e], d = dst[e];
  int hd = lane >> 3, f = lane*4;
  float sc = lrelu(asrc[s*4 + hd] + adst[d*4 + hd]);
  float e_ = expf(sc - o2f(m[d*4 + hd]));
  if ((lane & 7) == 0) atomicAdd(&den[d*4 + hd], e_);
  float4 x = *(const float4*)(xw + (size_t)s*HID + f);
  float* o = outb + (size_t)d*HID + f;
  atomicAdd(o + 0, e_*x.x); atomicAdd(o + 1, e_*x.y);
  atomicAdd(o + 2, e_*x.z); atomicAdd(o + 3, e_*x.w);
}

// ---------------- Set2Set ---------------------------------------------------
__global__ void k_lstm(const float* __restrict__ gates, float* cS, float* hS, int n){
  int t = blockIdx.x*256 + threadIdx.x; if (t >= n) return;
  int b = t / DJK, j = t % DJK;
  const float* gr = gates + (size_t)b*4*DJK;
  float i_ = sigf(gr[j]);
  float f_ = sigf(gr[DJK + j]);
  float g_ = tanhf(gr[2*DJK + j]);
  float o_ = sigf(gr[3*DJK + j]);
  float c = f_*cS[t] + i_*g_;
  cS[t] = c;
  hS[t] = o_*tanhf(c);
}
__global__ void k_ener(const float* __restrict__ x0, const float* __restrict__ x1,
                       const float* __restrict__ x2, const float* __restrict__ q,
                       const int* __restrict__ batch, float* __restrict__ ener,
                       unsigned* __restrict__ gmax){
  int t = blockIdx.x*256 + threadIdx.x;
  int node = t >> 5, lane = t & 31;
  if (node >= NN) return;
  int g = batch[node], f = lane*4;
  const float* qr = q + (size_t)g*DJK;
  float s = 0.f;
  { float4 a = *(const float4*)(x0 + (size_t)node*HID + f);
    float4 v = *(const float4*)(qr + f);
    s += a.x*v.x + a.y*v.y + a.z*v.z + a.w*v.w; }
  { float4 a = *(const float4*)(x1 + (size_t)node*HID + f);
    float4 v = *(const float4*)(qr + 128 + f);
    s += a.x*v.x + a.y*v.y + a.z*v.z + a.w*v.w; }
  { float4 a = *(const float4*)(x2 + (size_t)node*HID + f);
    float4 v = *(const float4*)(qr + 256 + f);
    s += a.x*v.x + a.y*v.y + a.z*v.z + a.w*v.w; }
#pragma unroll
  for (int msk = 1; msk < 32; msk <<= 1) s += __shfl_xor(s, msk);
  if (lane == 0){ ener[node] = s; atomicMax(&gmax[g], f2o(s)); }
}
__global__ void k_soft(const int* batch, const float* ener, const unsigned* gmax,
                       float* aexp, float* gsum){
  int t = blockIdx.x*256 + threadIdx.x; if (t >= NN) return;
  int g = batch[t];
  float a = expf(ener[t] - o2f(gmax[g]));
  aexp[t] = a;
  atomicAdd(&gsum[g], a);
}
__global__ void k_raccum(const float* __restrict__ x0, const float* __restrict__ x1,
                         const float* __restrict__ x2, const int* batch,
                         const float* aexp, const float* gsum, float* __restrict__ r){
  int t = blockIdx.x*256 + threadIdx.x;
  int node = t >> 5, lane = t & 31;
  if (node >= NN) return;
  int g = batch[node], f = lane*4;
  float w = aexp[node] / gsum[g];
  float* rp = r + (size_t)g*DJK;
  { float4 a = *(const float4*)(x0 + (size_t)node*HID + f);
    atomicAdd(rp + f + 0, w*a.x); atomicAdd(rp + f + 1, w*a.y);
    atomicAdd(rp + f + 2, w*a.z); atomicAdd(rp + f + 3, w*a.w); }
  { float4 a = *(const float4*)(x1 + (size_t)node*HID + f);
    atomicAdd(rp + 128 + f + 0, w*a.x); atomicAdd(rp + 128 + f + 1, w*a.y);
    atomicAdd(rp + 128 + f + 2, w*a.z); atomicAdd(rp + 128 + f + 3, w*a.w); }
  { float4 a = *(const float4*)(x2 + (size_t)node*HID + f);
    atomicAdd(rp + 256 + f + 0, w*a.x); atomicAdd(rp + 256 + f + 1, w*a.y);
    atomicAdd(rp + 256 + f + 2, w*a.z); atomicAdd(rp + 256 + f + 3, w*a.w); }
}
__global__ void k_qstar(const float* hS, const float* r, float* qs, int n){
  int t = blockIdx.x*256 + threadIdx.x; if (t >= n) return;
  int b = t / (2*DJK), j = t % (2*DJK);
  qs[t] = (j < DJK) ? hS[(size_t)b*DJK + j] : r[(size_t)b*DJK + (j - DJK)];
}

// ---------------------------------------------------------------------------
extern "C" void kernel_launch(void* const* d_in, const int* in_sizes, int n_in,
                              void* d_out, int out_size, void* d_ws, size_t ws_size,
                              hipStream_t stream){
  (void)in_sizes; (void)n_in; (void)out_size; (void)ws_size;
  const float* x      = (const float*)d_in[0];
  const int*   ei     = (const int*)  d_in[1];
  const float* eattr  = (const float*)d_in[2];
  const int*   batch  = (const int*)  d_in[3];
  const float* atom_W = (const float*)d_in[4];
  const float* atom_b = (const float*)d_in[5];
  struct Blk { const float *att_dst,*att_src,*eps,*gat_W,*gat_b,*ln1_b,*ln1_g,
                           *ln2_b,*ln2_g,*W1,*W2,*b1,*b2; } blk[3];
  for (int l = 0; l < 3; ++l){
    int o = 6 + l*13;
    blk[l].att_dst=(const float*)d_in[o+0];  blk[l].att_src=(const float*)d_in[o+1];
    blk[l].eps    =(const float*)d_in[o+2];  blk[l].gat_W  =(const float*)d_in[o+3];
    blk[l].gat_b  =(const float*)d_in[o+4];  blk[l].ln1_b  =(const float*)d_in[o+5];
    blk[l].ln1_g  =(const float*)d_in[o+6];  blk[l].ln2_b  =(const float*)d_in[o+7];
    blk[l].ln2_g  =(const float*)d_in[o+8];  blk[l].W1     =(const float*)d_in[o+9];
    blk[l].W2     =(const float*)d_in[o+10]; blk[l].b1     =(const float*)d_in[o+11];
    blk[l].b2     =(const float*)d_in[o+12];
  }
  const float* bond_W = (const float*)d_in[45];
  const float* bond_b = (const float*)d_in[46];
  const float* W_hh   = (const float*)d_in[47];
  const float* W_ih   = (const float*)d_in[48];
  const float* b_hh   = (const float*)d_in[49];
  const float* b_ih   = (const float*)d_in[50];
  const int* src = ei, *dst = ei + NE;

  // ---- workspace carve ----
  char* base = (char*)d_ws; size_t off = 0;
  auto carve = [&](size_t bytes)->void*{
    off = (off + 255) & ~(size_t)255;
    void* p = base + off; off += bytes; return p;
  };
  const size_t NB = (size_t)NN*HID*4;
  float*    ea    = (float*)   carve((size_t)NE*HID*4);
  float*    h0    = (float*)   carve(NB);
  float*    xs0   = (float*)   carve(NB);
  float*    xs1   = (float*)   carve(NB);
  float*    xs2   = (float*)   carve(NB);
  float*    bufA  = (float*)   carve(NB);
  float*    bufB  = (float*)   carve(NB);
  float*    bufC  = (float*)   carve(NB);
  float*    asrc  = (float*)   carve((size_t)NN*4*4);
  float*    adst  = (float*)   carve((size_t)NN*4*4);
  unsigned* mmax  = (unsigned*)carve((size_t)NN*4*4);
  float*    den   = (float*)   carve((size_t)NN*4*4);
  float*    gates = (float*)   carve((size_t)NG*4*DJK*4);
  float*    hS    = (float*)   carve((size_t)NG*DJK*4);
  float*    cS    = (float*)   carve((size_t)NG*DJK*4);
  float*    rbuf  = (float*)   carve((size_t)NG*DJK*4);
  float*    qs    = (float*)   carve((size_t)NG*2*DJK*4);
  float*    ener  = (float*)   carve((size_t)NN*4);
  float*    aexp  = (float*)   carve((size_t)NN*4);
  unsigned* gmax  = (unsigned*)carve((size_t)NG*4);
  float*    gsum  = (float*)   carve((size_t)NG*4);
  float*    biasc = (float*)   carve((size_t)4*DJK*4);
  _Float16* atomWf = (_Float16*)carve((size_t)2*8*512*2);
  _Float16* bondWf = (_Float16*)carve((size_t)1*8*512*2);
  _Float16 *W1f[3], *W2f[3], *gWf[3];
  for (int l = 0; l < 3; ++l){
    W1f[l] = (_Float16*)carve((size_t)4*8*512*2);
    W2f[l] = (_Float16*)carve((size_t)4*8*512*2);
    gWf[l] = (_Float16*)carve((size_t)4*8*512*2);
  }
  _Float16* Wihf = (_Float16*)carve((size_t)(768/32)*(1536/16)*512*2);
  _Float16* Whhf = (_Float16*)carve((size_t)(384/32)*(1536/16)*512*2);

  auto pack = [&](const float* W, int K, int NC, int KP, int ld, int tr, _Float16* f){
    int total = (KP/32)*(NC/16)*512;
    k_pack_bfrag<<<(total+255)/256, 256, 0, stream>>>(W, K, NC, KP, ld, tr, f, total);
  };
  auto gemm = [&](const float* A, int lda, int R, int K, int KP, const _Float16* Bf,
                  int NC, const float* bias, float* C, int ldc, int act, int accum){
    dim3 g((R+63)/64, NC/128);
    k_wmma_gemm<<<g, 256, 0, stream>>>(A, lda, R, K, KP, Bf, NC, bias, C, ldc, act, accum);
  };

  // ---- pack weights to f16 WMMA fragments (cheap, once per call) ----
  pack(atom_W, 64, 128, 64, 128, 0, atomWf);
  pack(bond_W, 16, 128, 32, 128, 0, bondWf);
  for (int l = 0; l < 3; ++l){
    pack(blk[l].W1,    128, 128, 128, 128, 0, W1f[l]);
    pack(blk[l].W2,    128, 128, 128, 128, 0, W2f[l]);
    pack(blk[l].gat_W, 128, 128, 128, 128, 0, gWf[l]);
  }
  pack(W_ih, 768, 1536, 768, 768, 1, Wihf);   // transposed: B = W_ih^T
  pack(W_hh, 384, 1536, 384, 384, 1, Whhf);
  k_addvec<<<(4*DJK+255)/256, 256, 0, stream>>>(b_ih, b_hh, biasc, 4*DJK);

  // ---- embeddings ----
  gemm(x,     64, NN,  64,  64, atomWf, 128, atom_b, h0, 128, 1, 0);
  gemm(eattr, 16, NE,  16,  32, bondWf, 128, bond_b, ea, 128, 1, 0);

  // ---- 3 GNN layers ----
  const int nNH = NN*HID;
  const float* hin = h0;
  float* xsl[3] = { xs0, xs1, xs2 };
  for (int l = 0; l < 3; ++l){
    k_scale<<<(nNH+255)/256, 256, 0, stream>>>(hin, blk[l].eps, bufA, nNH);
    k_gine_msg<<<(NE*32+255)/256, 256, 0, stream>>>(src, dst, hin, ea, bufA);
    gemm(bufA, 128, NN, 128, 128, W1f[l], 128, blk[l].b1, bufB, 128, 1, 0);
    gemm(bufB, 128, NN, 128, 128, W2f[l], 128, blk[l].b2, bufC, 128, 0, 0);
    k_ln<<<(NN*32+255)/256, 256, 0, stream>>>(bufC, nullptr, nullptr,
        blk[l].ln1_g, blk[l].ln1_b, hin, bufB, NN);            // bufB = g1
    gemm(bufB, 128, NN, 128, 128, gWf[l], 128, nullptr, bufA, 128, 0, 0); // bufA = xw
    k_att<<<(NN*32+255)/256, 256, 0, stream>>>(bufA, blk[l].att_src, blk[l].att_dst,
                                               asrc, adst);
    k_gat_maxinit<<<(NN*4+255)/256, 256, 0, stream>>>(asrc, adst, mmax, NN*4);
    k_gat_edge_max<<<(NE*4+255)/256, 256, 0, stream>>>(src, dst, asrc, adst, mmax);
    k_gat_self<<<(NN*32+255)/256, 256, 0, stream>>>(bufA, asrc, adst, mmax, den, bufC);
    k_gat_edge_acc<<<(NE*32+255)/256, 256, 0, stream>>>(src, dst, asrc, adst, mmax,
                                                        bufA, den, bufC);
    k_ln<<<(NN*32+255)/256, 256, 0, stream>>>(bufC, den, blk[l].gat_b,
        blk[l].ln2_g, blk[l].ln2_b, bufB, xsl[l], NN);
    hin = xsl[l];
  }

  // ---- Set2Set (3 steps) ----
  k_fill_f32<<<(NG*DJK+255)/256, 256, 0, stream>>>(hS, 0.f, NG*DJK);
  k_fill_f32<<<(NG*DJK+255)/256, 256, 0, stream>>>(cS, 0.f, NG*DJK);
  k_fill_f32<<<(NG*2*DJK+255)/256, 256, 0, stream>>>(qs, 0.f, NG*2*DJK);
  const unsigned NEG_INF_ORD = 0x007FFFFFu;   // f2o(-inf)
  for (int s = 0; s < 3; ++s){
    gemm(qs, 2*DJK, NG, 2*DJK, 2*DJK, Wihf, 4*DJK, biasc,   gates, 4*DJK, 0, 0);
    gemm(hS,   DJK, NG,   DJK,   DJK, Whhf, 4*DJK, nullptr, gates, 4*DJK, 0, 1);
    k_lstm<<<(NG*DJK+255)/256, 256, 0, stream>>>(gates, cS, hS, NG*DJK);
    k_fill_u32<<<(NG+255)/256, 256, 0, stream>>>(gmax, NEG_INF_ORD, NG);
    k_ener<<<(NN*32+255)/256, 256, 0, stream>>>(xs0, xs1, xs2, hS, batch, ener, gmax);
    k_fill_f32<<<(NG+255)/256, 256, 0, stream>>>(gsum, 0.f, NG);
    k_soft<<<(NN+255)/256, 256, 0, stream>>>(batch, ener, gmax, aexp, gsum);
    k_fill_f32<<<(NG*DJK+255)/256, 256, 0, stream>>>(rbuf, 0.f, NG*DJK);
    k_raccum<<<(NN*32+255)/256, 256, 0, stream>>>(xs0, xs1, xs2, batch, aexp, gsum, rbuf);
    k_qstar<<<(NG*2*DJK+255)/256, 256, 0, stream>>>(hS, rbuf, qs, NG*2*DJK);
  }
  k_copy<<<(NG*2*DJK+255)/256, 256, 0, stream>>>(qs, (float*)d_out, NG*2*DJK);
}
